// SpatialTemporalSelfAttention_23630910063009
// MI455X (gfx1250) — matrix operand order repaired and checked by reading, not compiled, not run on hardware
//
#include <hip/hip_runtime.h>
#include <hip/hip_bf16.h>
#include <math.h>

// ---------------- problem constants ----------------
#define BB 8
#define SS 2048
#define DD 512
#define BSD (BB * SS * DD)        // 8,388,608
#define WSZ (DD * DD)             // 262,144
#define RADF 0.017453292519943295f
#define RADHF 0.008726646259971648f
#define SCALEF 0.044194173824159216f   // 1/sqrt(512)
#define EARTH_R 6371.0f

typedef __attribute__((ext_vector_type(16))) __bf16 v16bf;
typedef __attribute__((ext_vector_type(8)))  float  v8f;

union Frag16 {
    v16bf v;
    uint4 q[2];
};

__device__ __forceinline__ unsigned short f2bf(float f) {
    union { float f; unsigned int u; } x;
    x.f = f;
    unsigned int u = x.u;
    unsigned int r = u + 0x7fffu + ((u >> 16) & 1u);   // round-to-nearest-even
    return (unsigned short)(r >> 16);
}

// ---------------------------------------------------------------
// Kernel 1: f32 -> bf16 conversion of X and the three weight mats
// ---------------------------------------------------------------
__global__ void cvt_kernel(const float* __restrict__ X,
                           const float* __restrict__ Wq,
                           const float* __restrict__ Wk,
                           const float* __restrict__ Wv,
                           unsigned short* __restrict__ Xb,
                           unsigned short* __restrict__ Wqb,
                           unsigned short* __restrict__ Wkb,
                           unsigned short* __restrict__ Wvb) {
    long long i = (long long)blockIdx.x * blockDim.x + threadIdx.x;
    if (i < BSD) {
        Xb[i] = f2bf(X[i]);
    } else {
        long long j = i - BSD;
        if (j < WSZ)           Wqb[j]            = f2bf(Wq[j]);
        else if (j < 2 * WSZ)  Wkb[j - WSZ]      = f2bf(Wk[j - WSZ]);
        else if (j < 3 * WSZ)  Wvb[j - 2 * WSZ]  = f2bf(Wv[j - 2 * WSZ]);
    }
}

// ---------------------------------------------------------------
// Kernel 2: QKV projection. y = x @ W^T + b via WMMA bf16.
// One wave computes one 16x16 output tile (K = 512 -> 16 wmma).
// Q,K stored row-major bf16; V stored transposed per batch [D][S].
// ---------------------------------------------------------------
__global__ __launch_bounds__(128)
void proj_kernel(const unsigned short* __restrict__ Xb,
                 const unsigned short* __restrict__ Wqb,
                 const unsigned short* __restrict__ Wkb,
                 const unsigned short* __restrict__ Wvb,
                 const float* __restrict__ bq,
                 const float* __restrict__ bk,
                 const float* __restrict__ bv,
                 unsigned short* __restrict__ Qb,
                 unsigned short* __restrict__ Kb,
                 unsigned short* __restrict__ Vtb) {
    const int lane = threadIdx.x & 31;
    const int wave = (blockIdx.x * (blockDim.x >> 5)) + (threadIdx.x >> 5);
    // total waves = 3 * (BS/16) * (D/16) = 3 * 1024 * 32
    const int which = wave >> 15;              // / 32768
    const int rem   = wave & 32767;
    const int mt    = rem >> 5;                // / 32  (row tile, 0..1023)
    const int nt    = rem & 31;                // col tile, 0..31

    const unsigned short* W = (which == 0) ? Wqb : (which == 1) ? Wkb : Wvb;
    const float* bias       = (which == 0) ? bq  : (which == 1) ? bk  : bv;

    const int m = lane & 15;   // row (A) / col (B,C)
    const int g = lane >> 4;   // half-wave group

    v8f acc;
    #pragma unroll
    for (int r = 0; r < 8; ++r) acc[r] = 0.0f;

    const unsigned short* arow = Xb + (size_t)(mt * 16 + m) * DD;
    const unsigned short* brow = W  + (size_t)(nt * 16 + m) * DD;

    #pragma unroll 4
    for (int kk = 0; kk < 16; ++kk) {
        Frag16 a, bfr;
        const unsigned short* ap = arow + kk * 32;
        a.q[0] = *(const uint4*)(ap + g * 8);
        a.q[1] = *(const uint4*)(ap + 16 + g * 8);
        const unsigned short* bp = brow + kk * 32;
        bfr.q[0] = *(const uint4*)(bp + g * 8);
        bfr.q[1] = *(const uint4*)(bp + 16 + g * 8);
        acc = __builtin_amdgcn_wmma_f32_16x16x32_bf16(
                  false, a.v, false, bfr.v, (short)0, acc, false, false);
    }

    const float bval = bias[nt * 16 + m];
    const int col = nt * 16 + m;

    #pragma unroll
    for (int r = 0; r < 8; ++r) {
        const float y = acc[r] + bval;
        const int grow = mt * 16 + r + 8 * g;          // global row in [0,BS)
        if (which == 2) {
            const int b = grow >> 11;                  // / S
            const int s = grow & (SS - 1);
            Vtb[(size_t)b * DD * SS + (size_t)col * SS + s] = f2bf(y);
        } else {
            unsigned short* dst = (which == 0) ? Qb : Kb;
            dst[(size_t)grow * DD + col] = f2bf(y);
        }
    }
}

// ---------------------------------------------------------------
// Kernel 3: fused flash attention with spatio-temporal bias.
// 1 block = (batch b, 16-query tile). 8 waves, 256 threads.
// Q A-fragments are register-resident for the whole kernel
// (16 k-steps x 8 VGPRs = 128 VGPRs); no LDS staging of Q.
// ---------------------------------------------------------------
__global__ __launch_bounds__(256)
void flash_kernel(const unsigned short* __restrict__ Qb,
                  const unsigned short* __restrict__ Kb,
                  const unsigned short* __restrict__ Vtb,
                  const float* __restrict__ tseq,
                  const float* __restrict__ lat,
                  const float* __restrict__ lon,
                  const int*   __restrict__ mask,
                  float* __restrict__ out) {
    __shared__ float          Sf[16 * 128];    // 8 KB
    __shared__ unsigned short Pb[16 * 128];    // 4 KB
    __shared__ float mrow[16], lrow[16], arow[16];

    const int b  = blockIdx.x >> 7;            // 128 query tiles per batch
    const int qt = blockIdx.x & 127;
    const int i0 = qt * 16;
    const int tid  = threadIdx.x;
    const int lane = tid & 31;
    const int w    = tid >> 5;
    const int m    = lane & 15;
    const int g    = lane >> 4;

    if (tid < 16) { mrow[tid] = -INFINITY; lrow[tid] = 0.0f; }

    // ---- hoist all Q A-fragments into registers (loaded once) ----
    Frag16 qa[16];
    {
        const unsigned short* qrow = Qb + ((size_t)b * SS + i0 + m) * DD;
        #pragma unroll
        for (int kk = 0; kk < 16; ++kk) {
            const unsigned short* ap = qrow + kk * 32;
            qa[kk].q[0] = *(const uint4*)(ap + g * 8);
            qa[kk].q[1] = *(const uint4*)(ap + 16 + g * 8);
        }
    }

    // per-lane query-row stats (C layout rows: r + 8*g)
    float t_i[8], lath_i[8], lonh_i[8], cos_i[8];
    #pragma unroll
    for (int r = 0; r < 8; ++r) {
        const int ii = b * SS + i0 + r + 8 * g;
        t_i[r]    = tseq[ii];
        const float la = lat[ii];
        lath_i[r] = la * RADHF;
        lonh_i[r] = lon[ii] * RADHF;
        cos_i[r]  = __cosf(la * RADF);
    }

    v8f acc[4];
    #pragma unroll
    for (int n = 0; n < 4; ++n)
        #pragma unroll
        for (int r = 0; r < 8; ++r) acc[n][r] = 0.0f;

    for (int jt = 0; jt < 16; ++jt) {
        const int j0 = jt * 128;
        const int c0 = j0 + w * 16;            // this wave's 16 score columns

        // ---- S = Q @ K^T for 16x16 tile, K-dim = 512 ----
        v8f s;
        #pragma unroll
        for (int r = 0; r < 8; ++r) s[r] = 0.0f;
        const unsigned short* krow = Kb + ((size_t)b * SS + c0 + m) * DD;
        #pragma unroll
        for (int kk = 0; kk < 16; ++kk) {
            Frag16 bf;
            const unsigned short* bp = krow + kk * 32;
            bf.q[0] = *(const uint4*)(bp + g * 8);
            bf.q[1] = *(const uint4*)(bp + 16 + g * 8);
            s = __builtin_amdgcn_wmma_f32_16x16x32_bf16(
                    false, qa[kk].v, false, bf.v, (short)0, s, false, false);
        }

        // ---- bias: scale - haversine - |dt| + mask ----
        {
            const int jj = b * SS + c0 + m;
            const float tj    = tseq[jj];
            const float laj   = lat[jj];
            const float lathj = laj * RADHF;
            const float lonhj = lon[jj] * RADHF;
            const float cosj  = __cosf(laj * RADF);
            const float mk    = (mask[jj] == 0) ? -1e9f : 0.0f;
            #pragma unroll
            for (int r = 0; r < 8; ++r) {
                const float dt    = fabsf(tj - t_i[r]);
                const float sdla  = __sinf(lathj - lath_i[r]);
                const float sdlo  = __sinf(lonhj - lonh_i[r]);
                float a_h = sdla * sdla + cos_i[r] * cosj * sdlo * sdlo;
                a_h = fminf(fmaxf(a_h, 0.0f), 1.0f);
                const float c = 2.0f * atan2f(sqrtf(a_h), sqrtf(1.0f - a_h));
                Sf[(r + 8 * g) * 128 + w * 16 + m] =
                    s[r] * SCALEF - EARTH_R * c - dt + mk;
            }
        }
        __syncthreads();

        // ---- online softmax: wave w owns rows 2w, 2w+1 ----
        #pragma unroll
        for (int rr = 0; rr < 2; ++rr) {
            const int row = 2 * w + rr;
            float v0 = Sf[row * 128 + lane];
            float v1 = Sf[row * 128 + lane + 32];
            float v2 = Sf[row * 128 + lane + 64];
            float v3 = Sf[row * 128 + lane + 96];
            float bmax = fmaxf(fmaxf(v0, v1), fmaxf(v2, v3));
            #pragma unroll
            for (int o = 16; o > 0; o >>= 1)
                bmax = fmaxf(bmax, __shfl_xor(bmax, o, 32));
            const float mprev = mrow[row];
            const float mnew  = fmaxf(mprev, bmax);
            const float e0 = __expf(v0 - mnew);
            const float e1 = __expf(v1 - mnew);
            const float e2 = __expf(v2 - mnew);
            const float e3 = __expf(v3 - mnew);
            Pb[row * 128 + lane]      = f2bf(e0);
            Pb[row * 128 + lane + 32] = f2bf(e1);
            Pb[row * 128 + lane + 64] = f2bf(e2);
            Pb[row * 128 + lane + 96] = f2bf(e3);
            float bsum = (e0 + e1) + (e2 + e3);
            #pragma unroll
            for (int o = 16; o > 0; o >>= 1)
                bsum += __shfl_xor(bsum, o, 32);
            if (lane == 0) {
                const float al = __expf(mprev - mnew);
                mrow[row] = mnew;
                lrow[row] = al * lrow[row] + bsum;
                arow[row] = al;
            }
        }
        __syncthreads();

        // ---- rescale accumulators, then P @ V (wave's 64 output cols) ----
        float alr[8];
        #pragma unroll
        for (int r = 0; r < 8; ++r) alr[r] = arow[r + 8 * g];
        #pragma unroll
        for (int n = 0; n < 4; ++n)
            #pragma unroll
            for (int r = 0; r < 8; ++r) acc[n][r] *= alr[r];

        Frag16 afr[4];
        #pragma unroll
        for (int kk = 0; kk < 4; ++kk) {
            const unsigned short* ap = Pb + m * 128 + kk * 32;
            afr[kk].q[0] = *(const uint4*)(ap + g * 8);
            afr[kk].q[1] = *(const uint4*)(ap + 16 + g * 8);
        }
        #pragma unroll
        for (int n = 0; n < 4; ++n) {
            const int e0 = w * 64 + n * 16 + m;
            const unsigned short* vcol =
                Vtb + (size_t)b * DD * SS + (size_t)e0 * SS + j0;
            #pragma unroll
            for (int kk = 0; kk < 4; ++kk) {
                Frag16 bf;
                const unsigned short* bp = vcol + kk * 32;
                bf.q[0] = *(const uint4*)(bp + g * 8);
                bf.q[1] = *(const uint4*)(bp + 16 + g * 8);
                acc[n] = __builtin_amdgcn_wmma_f32_16x16x32_bf16(
                             false, afr[kk].v, false, bf.v, (short)0,
                             acc[n], false, false);
            }
        }
        __syncthreads();
    }

    // ---- epilogue: divide by l, store f32 ----
    float linv[8];
    #pragma unroll
    for (int r = 0; r < 8; ++r) linv[r] = 1.0f / lrow[r + 8 * g];
    #pragma unroll
    for (int n = 0; n < 4; ++n) {
        const int col = w * 64 + n * 16 + m;
        #pragma unroll
        for (int r = 0; r < 8; ++r) {
            const int row = i0 + r + 8 * g;
            out[((size_t)b * SS + row) * DD + col] = acc[n][r] * linv[r];
        }
    }
}

// ---------------------------------------------------------------
// Host launcher
// ---------------------------------------------------------------
extern "C" void kernel_launch(void* const* d_in, const int* in_sizes, int n_in,
                              void* d_out, int out_size, void* d_ws, size_t ws_size,
                              hipStream_t stream) {
    const float* X    = (const float*)d_in[0];
    const float* tseq = (const float*)d_in[1];
    const float* lat  = (const float*)d_in[2];
    const float* lon  = (const float*)d_in[3];
    const int*   mask = (const int*)  d_in[4];
    const float* Wq   = (const float*)d_in[5];
    const float* bq   = (const float*)d_in[6];
    const float* Wk   = (const float*)d_in[7];
    const float* bk   = (const float*)d_in[8];
    const float* Wv   = (const float*)d_in[9];
    const float* bv   = (const float*)d_in[10];
    float* out = (float*)d_out;

    char* ws = (char*)d_ws;
    size_t off = 0;
    unsigned short* Xb  = (unsigned short*)(ws + off); off += (size_t)BSD * 2;
    unsigned short* Wqb = (unsigned short*)(ws + off); off += (size_t)WSZ * 2;
    unsigned short* Wkb = (unsigned short*)(ws + off); off += (size_t)WSZ * 2;
    unsigned short* Wvb = (unsigned short*)(ws + off); off += (size_t)WSZ * 2;
    unsigned short* Qb  = (unsigned short*)(ws + off); off += (size_t)BSD * 2;
    unsigned short* Kb  = (unsigned short*)(ws + off); off += (size_t)BSD * 2;
    unsigned short* Vtb = (unsigned short*)(ws + off); off += (size_t)BSD * 2;

    // 1) convert f32 -> bf16
    {
        const long long total = (long long)BSD + 3LL * WSZ;   // 9,175,040
        const int blocks = (int)((total + 255) / 256);
        cvt_kernel<<<blocks, 256, 0, stream>>>(X, Wq, Wk, Wv, Xb, Wqb, Wkb, Wvb);
    }
    // 2) QKV projection: 3 * 1024 * 32 waves, 4 waves / block
    {
        const int total_waves = 3 * 1024 * 32;
        proj_kernel<<<total_waves / 4, 128, 0, stream>>>(
            Xb, Wqb, Wkb, Wvb, bq, bk, bv, Qb, Kb, Vtb);
    }
    // 3) fused flash attention with spatio-temporal bias
    {
        flash_kernel<<<BB * (SS / 16), 256, 0, stream>>>(
            Qb, Kb, Vtb, tseq, lat, lon, mask, out);
    }
}
